// NERNetAttention_59038620451310
// MI455X (gfx1250) — matrix operand order repaired and checked by reading, not compiled, not run on hardware
//
#include <hip/hip_runtime.h>

// ---------------------------------------------------------------------------
// BiLSTM + self-attention NER head for MI455X (gfx1250, wave32, WMMA).
//   k_prep  : fp32->bf16 weight conversion + bias sums
//   k_embed : embedding gather -> bf16
//   k_proj  : X @ Wih^T + bias for all 16384 timesteps (WMMA bf16 GEMM), x2 dirs
//             gates stored [t][batch][512] so each step's block is contiguous
//   k_lstm  : persistent recurrence; Whh resident in LDS (128 KB of the WGP's
//             320 KB); h @ Whh^T per step via WMMA; prefetch next gate block
//   k_attn  : softmax(y y^T) y + FC + log_softmax, WMMA tiles + LDS scores
// ---------------------------------------------------------------------------

#define Bn 16
#define Sn 1024
#define En 128
#define Hn 128
#define Cn 32

typedef __attribute__((ext_vector_type(16))) __bf16        v16bf;
typedef __attribute__((ext_vector_type(8)))  float         v8f;
typedef __attribute__((ext_vector_type(4)))  unsigned int  v4u;

union Frag {
    v16bf          v;
    unsigned short u[16];
    v4u            q[2];
};

__device__ __forceinline__ unsigned short f2bf(float f) {
    unsigned int x = __float_as_uint(f);
    x = (x + 0x7FFFu + ((x >> 16) & 1u)) >> 16;   // round-to-nearest-even
    return (unsigned short)x;
}
__device__ __forceinline__ float bf2f(unsigned short h) {
    return __uint_as_float(((unsigned int)h) << 16);
}
__device__ __forceinline__ float sigm(float x) { return 1.0f / (1.0f + __expf(-x)); }

__device__ __forceinline__ v8f wmma_bf(const Frag& a, const Frag& b, v8f c) {
    return __builtin_amdgcn_wmma_f32_16x16x32_bf16(
        false, a.v, false, b.v, (short)0, c, false, false);
}

// --------------------------- weight prep -----------------------------------
__global__ void __launch_bounds__(256) k_prep(
    const float* __restrict__ wihf, const float* __restrict__ whhf,
    const float* __restrict__ wihb, const float* __restrict__ whhb,
    const float* __restrict__ bihf, const float* __restrict__ bhhf,
    const float* __restrict__ bihb, const float* __restrict__ bhhb,
    unsigned short* __restrict__ o_wihf, unsigned short* __restrict__ o_whhf,
    unsigned short* __restrict__ o_wihb, unsigned short* __restrict__ o_whhb,
    float* __restrict__ o_biasf, float* __restrict__ o_biasb)
{
    int i = blockIdx.x * 256 + threadIdx.x;           // 0 .. 262143
    int which = i >> 16;
    int off   = i & 65535;
    const float* src = (which == 0) ? wihf : (which == 1) ? whhf
                     : (which == 2) ? wihb : whhb;
    unsigned short* dst = (which == 0) ? o_wihf : (which == 1) ? o_whhf
                        : (which == 2) ? o_wihb : o_whhb;
    dst[off] = f2bf(src[off]);
    if (i < 512)        o_biasf[i]       = bihf[i] + bhhf[i];
    else if (i < 1024)  o_biasb[i - 512] = bihb[i - 512] + bhhb[i - 512];
}

// --------------------------- embedding gather ------------------------------
__global__ void __launch_bounds__(256) k_embed(
    const int* __restrict__ tok, const float* __restrict__ emb,
    unsigned short* __restrict__ exb)
{
    long i = (long)blockIdx.x * 256 + threadIdx.x;    // B*S*E threads exactly
    int  e = (int)(i & (En - 1));
    long n = i >> 7;
    int  t = tok[n];
    exb[i] = f2bf(emb[(long)t * En + e]);
}

// ------------------ input projection: G = X @ Wih^T + bias -----------------
// X: [16384 x 128] bf16 row-major,  W: [512 x 128] bf16 row-major (NT GEMM).
// Output layout: gates[(s*16 + b)*512 + j]  (per-timestep contiguous blocks).
__global__ void __launch_bounds__(256) k_proj(
    const unsigned short* __restrict__ exb,
    const unsigned short* __restrict__ wbf,
    const float* __restrict__ bias,
    float* __restrict__ gates)
{
    int tid = threadIdx.x;
    int wv = tid >> 5, lane = tid & 31, side = lane >> 4, lp = lane & 15;
    long mt = blockIdx.x;                              // 1024 row tiles
    const unsigned short* arow = exb + (mt * 16 + lp) * En;

    Frag a[4];                                         // A frags reused by all tiles
#pragma unroll
    for (int ks = 0; ks < 4; ++ks) {
        a[ks].q[0] = *(const v4u*)(arow + ks * 32 + side * 8);
        a[ks].q[1] = *(const v4u*)(arow + ks * 32 + 16 + side * 8);
    }

    for (int i = 0; i < 4; ++i) {
        int nt   = wv * 4 + i;                         // 0..31 N-tiles
        int ncol = nt * 16 + lp;
        float bv = bias[ncol];
        v8f acc;
#pragma unroll
        for (int r = 0; r < 8; ++r) acc[r] = bv;
        const unsigned short* brow = wbf + (long)ncol * En;
        Frag b[4];                                     // batch all B loads
#pragma unroll
        for (int ks = 0; ks < 4; ++ks) {
            b[ks].q[0] = *(const v4u*)(brow + ks * 32 + side * 8);
            b[ks].q[1] = *(const v4u*)(brow + ks * 32 + 16 + side * 8);
        }
#pragma unroll
        for (int ks = 0; ks < 4; ++ks) acc = wmma_bf(a[ks], b[ks], acc);

        int n0 = (int)(mt * 16) + side * 8;
#pragma unroll
        for (int r = 0; r < 8; ++r) {                  // row n -> (b = n>>10, s = n&1023)
            int n = n0 + r;
            int bb = n >> 10, ss = n & 1023;
            gates[((long)ss * Bn + bb) * 512 + ncol] = acc[r];
        }
    }
}

// ------------------------- LSTM recurrence ---------------------------------
// One persistent 256-thread block per direction; Whh lives in LDS. Wave w owns
// hidden units [16w,16w+16) so its 4 gate tiles share the (lane,reg)->(b,u) map.
__global__ void __launch_bounds__(256) k_lstm(
    const float* __restrict__ gates_f, const float* __restrict__ gates_b,
    const unsigned short* __restrict__ whh_f,
    const unsigned short* __restrict__ whh_b,
    const int* __restrict__ xlen,
    unsigned short* __restrict__ ybf, unsigned short* __restrict__ ybfT)
{
    __shared__ __attribute__((aligned(16))) unsigned short sW[512 * Hn];  // 128 KB
    __shared__ __attribute__((aligned(16))) float          sh[Bn * Hn];
    __shared__ __attribute__((aligned(16))) float          sc[Bn * Hn];
    __shared__ __attribute__((aligned(16))) unsigned short shb[Bn * Hn];
    __shared__ int slen[Bn];

    int dir = blockIdx.x;
    const float*          gx  = dir ? gates_b : gates_f;
    const unsigned short* whh = dir ? whh_b   : whh_f;

    int tid = threadIdx.x;
    {   // stage Whh into LDS once (8192 x b128)
        const v4u* src = (const v4u*)whh;
        v4u*       dst = (v4u*)sW;
        for (int i = tid; i < (512 * Hn) / 8; i += 256) dst[i] = src[i];
    }
    for (int i = tid; i < Bn * Hn; i += 256) { sh[i] = 0.f; sc[i] = 0.f; shb[i] = 0; }
    if (tid < Bn) slen[tid] = xlen[tid];
    __syncthreads();

    int wv = tid >> 5, lane = tid & 31, side = lane >> 4, lp = lane & 15;
    int u = wv * 16 + lp;                              // hidden unit column

    for (int t = 0; t < Sn; ++t) {
        int tt = dir ? (Sn - 1 - t) : t;

        // prefetch next step's 32 KB gate block (one cacheline per thread)
        int tn = dir ? (tt - 1) : (tt + 1);
        if (tn >= 0 && tn < Sn)
            __builtin_prefetch((const char*)(gx + (long)tn * Bn * 512) + tid * 128, 0, 1);

        Frag a[4];                                     // h frags shared by 4 gates
#pragma unroll
        for (int ks = 0; ks < 4; ++ks) {
            a[ks].q[0] = *(const v4u*)(shb + lp * Hn + ks * 32 + side * 8);
            a[ks].q[1] = *(const v4u*)(shb + lp * Hn + ks * 32 + 16 + side * 8);
        }

        v8f acc[4];
#pragma unroll
        for (int g = 0; g < 4; ++g) {
            int ncol = g * Hn + u;
#pragma unroll
            for (int r = 0; r < 8; ++r) {              // C init = precomputed x-gates
                int m = side * 8 + r;
                acc[g][r] = gx[((long)tt * Bn + m) * 512 + ncol];
            }
            const unsigned short* brow = sW + (long)ncol * Hn;
            Frag b[4];
#pragma unroll
            for (int ks = 0; ks < 4; ++ks) {
                b[ks].q[0] = *(const v4u*)(brow + ks * 32 + side * 8);
                b[ks].q[1] = *(const v4u*)(brow + ks * 32 + 16 + side * 8);
            }
#pragma unroll
            for (int ks = 0; ks < 4; ++ks) acc[g] = wmma_bf(a[ks], b[ks], acc[g]);
        }
        __syncthreads();                               // all shb reads done
#pragma unroll
        for (int r = 0; r < 8; ++r) {
            int m = side * 8 + r;
            float gi = sigm(acc[0][r]);
            float gf = sigm(acc[1][r]);
            float gg = tanhf(acc[2][r]);
            float go = sigm(acc[3][r]);
            float co = sc[m * Hn + u], ho = sh[m * Hn + u];
            float cn = gf * co + gi * gg;
            float hn = go * tanhf(cn);
            float msk = (tt < slen[m]) ? 1.0f : 0.0f;
            float h2 = msk * hn + (1.0f - msk) * ho;
            float c2 = msk * cn + (1.0f - msk) * co;
            float ov = msk * hn;
            sh[m * Hn + u] = h2;  sc[m * Hn + u] = c2;  shb[m * Hn + u] = f2bf(h2);
            int d = dir * Hn + u;
            ybf [((long)m * Sn + tt) * 256 + d] = f2bf(ov);
            ybfT[((long)m * 256 + d) * Sn + tt] = f2bf(ov);
        }
        __syncthreads();                               // state visible next step
    }
}

// ------------------- attention + FC + log_softmax --------------------------
__global__ void __launch_bounds__(256) k_attn(
    const unsigned short* __restrict__ ybf,
    const unsigned short* __restrict__ ybfT,
    const float* __restrict__ fcW, const float* __restrict__ fcb,
    float* __restrict__ out)
{
    __shared__ __attribute__((aligned(16))) unsigned short sQ[16 * 256];   // 8 KB
    __shared__ __attribute__((aligned(16))) unsigned short sS[16 * 1024];  // 32 KB
    __shared__ __attribute__((aligned(16))) float          sO[16 * 256];   // 16 KB
    __shared__ float sRed[16 * 16];
    __shared__ float sMax[16];
    __shared__ float sInv[16];
    __shared__ float sLog[16 * 32];
    __shared__ float sLse[16];

    int b  = blockIdx.x >> 6;
    int mt = blockIdx.x & 63;
    int tid = threadIdx.x;

    {   // stage Q tile (16 x 256 bf16) as b128 copies
        const v4u* src = (const v4u*)(ybf + ((long)b * Sn + mt * 16) * 256);
        v4u* dst = (v4u*)sQ;
        for (int i = tid; i < 512; i += 256) dst[i] = src[i];
    }
    __syncthreads();

    int wv = tid >> 5, lane = tid & 31, side = lane >> 4, lp = lane & 15;

    // ---- scores: S = Q @ K^T, 64 column tiles over 8 waves ----
    {
        Frag qa[8];                                    // Q frags reused by all tiles
#pragma unroll
        for (int ks = 0; ks < 8; ++ks) {
            qa[ks].q[0] = *(const v4u*)(sQ + lp * 256 + ks * 32 + side * 8);
            qa[ks].q[1] = *(const v4u*)(sQ + lp * 256 + ks * 32 + 16 + side * 8);
        }
        for (int ct = wv; ct < 64; ct += 8) {
            v8f acc = {};
            const unsigned short* kb = ybf + ((long)b * Sn + ct * 16 + lp) * 256;
#pragma unroll
            for (int ks = 0; ks < 8; ++ks) {           // K = 256
                Frag bb;
                bb.q[0] = *(const v4u*)(kb + ks * 32 + side * 8);
                bb.q[1] = *(const v4u*)(kb + ks * 32 + 16 + side * 8);
                acc = wmma_bf(qa[ks], bb, acc);
            }
#pragma unroll
            for (int r = 0; r < 8; ++r)
                sS[(side * 8 + r) * 1024 + ct * 16 + lp] = f2bf(acc[r]);
        }
    }
    __syncthreads();

    // ---- softmax per row (16 threads/row, 64 cols each) ----
    int row = tid >> 4, seg = tid & 15;
    {
        float mx = -1e30f;
        for (int j = seg * 64; j < seg * 64 + 64; ++j)
            mx = fmaxf(mx, bf2f(sS[row * 1024 + j]));
        sRed[row * 16 + seg] = mx;
    }
    __syncthreads();
    if (seg == 0) {
        float mx = sRed[row * 16];
        for (int k = 1; k < 16; ++k) mx = fmaxf(mx, sRed[row * 16 + k]);
        sMax[row] = mx;
    }
    __syncthreads();
    {
        float mx = sMax[row], s = 0.f;
        for (int j = seg * 64; j < seg * 64 + 64; ++j)
            s += __expf(bf2f(sS[row * 1024 + j]) - mx);
        sRed[row * 16 + seg] = s;
    }
    __syncthreads();
    if (seg == 0) {
        float s = 0.f;
        for (int k = 0; k < 16; ++k) s += sRed[row * 16 + k];
        sInv[row] = 1.0f / s;
    }
    __syncthreads();
    {
        float mx = sMax[row], inv = sInv[row];
        for (int j = seg * 64; j < seg * 64 + 64; ++j) {
            int idx = row * 1024 + j;
            sS[idx] = f2bf(__expf(bf2f(sS[idx]) - mx) * inv);
        }
    }
    __syncthreads();

    // ---- O = P @ V : 16 output N-tiles over 8 waves, K = 1024 ----
    for (int nt = wv; nt < 16; nt += 8) {
        v8f acc = {};
        const unsigned short* vb = ybfT + ((long)b * 256 + nt * 16 + lp) * Sn;
        for (int ks = 0; ks < 32; ++ks) {
            Frag a, bb;
            a.q[0]  = *(const v4u*)(sS + lp * 1024 + ks * 32 + side * 8);
            a.q[1]  = *(const v4u*)(sS + lp * 1024 + ks * 32 + 16 + side * 8);
            bb.q[0] = *(const v4u*)(vb + ks * 32 + side * 8);
            bb.q[1] = *(const v4u*)(vb + ks * 32 + 16 + side * 8);
            acc = wmma_bf(a, bb, acc);
        }
#pragma unroll
        for (int r = 0; r < 8; ++r)
            sO[(side * 8 + r) * 256 + nt * 16 + lp] = acc[r];
    }
    __syncthreads();

    // ---- fused FC (256 -> 32) + log_softmax epilogue ----
    {
        int r2 = tid >> 4, cb = tid & 15;
        for (int cc = 0; cc < 2; ++cc) {
            int cls = cb + cc * 16;
            const float* wrow = fcW + cls * 256;
            float s = fcb[cls];
            for (int d = 0; d < 256; ++d) s += sO[r2 * 256 + d] * wrow[d];
            sLog[r2 * 32 + cls] = s;
        }
    }
    __syncthreads();
    if (tid < 16) {
        float mx = -1e30f;
        for (int c = 0; c < Cn; ++c) mx = fmaxf(mx, sLog[tid * 32 + c]);
        float s = 0.f;
        for (int c = 0; c < Cn; ++c) s += __expf(sLog[tid * 32 + c] - mx);
        sLse[tid] = mx + __logf(s);
    }
    __syncthreads();
    {
        int r2 = tid >> 4, cb = tid & 15;
        long ob = ((long)b * Sn + mt * 16 + r2) * Cn;
        for (int cc = 0; cc < 2; ++cc) {
            int cls = cb + cc * 16;
            out[ob + cls] = sLog[r2 * 32 + cls] - sLse[r2];
        }
    }
}

// ---------------------------------------------------------------------------
extern "C" void kernel_launch(void* const* d_in, const int* in_sizes, int n_in,
                              void* d_out, int out_size, void* d_ws, size_t ws_size,
                              hipStream_t stream) {
    const int*   tokens = (const int*)  d_in[0];
    const int*   xlen   = (const int*)  d_in[1];
    const float* emb    = (const float*)d_in[2];
    const float* wihf   = (const float*)d_in[3];
    const float* whhf   = (const float*)d_in[4];
    const float* bihf   = (const float*)d_in[5];
    const float* bhhf   = (const float*)d_in[6];
    const float* wihb   = (const float*)d_in[7];
    const float* whhb   = (const float*)d_in[8];
    const float* bihb   = (const float*)d_in[9];
    const float* bhhb   = (const float*)d_in[10];
    const float* fcW    = (const float*)d_in[11];
    const float* fcb    = (const float*)d_in[12];
    float* out = (float*)d_out;

    char*  ws  = (char*)d_ws;
    size_t off = 0;
    auto carve = [&](size_t bytes) -> char* {
        char* p = ws + off;
        off += (bytes + 255) & ~(size_t)255;
        return p;
    };
    unsigned short* o_wihf = (unsigned short*)carve(512 * 128 * 2);
    unsigned short* o_whhf = (unsigned short*)carve(512 * 128 * 2);
    unsigned short* o_wihb = (unsigned short*)carve(512 * 128 * 2);
    unsigned short* o_whhb = (unsigned short*)carve(512 * 128 * 2);
    float*          biasf  = (float*)carve(512 * 4);
    float*          biasb  = (float*)carve(512 * 4);
    unsigned short* exb    = (unsigned short*)carve((size_t)Bn * Sn * En * 2);
    float*          gatesf = (float*)carve((size_t)Bn * Sn * 512 * 4);
    float*          gatesb = (float*)carve((size_t)Bn * Sn * 512 * 4);
    unsigned short* ybf    = (unsigned short*)carve((size_t)Bn * Sn * 256 * 2);
    unsigned short* ybfT   = (unsigned short*)carve((size_t)Bn * Sn * 256 * 2);

    k_prep<<<1024, 256, 0, stream>>>(wihf, whhf, wihb, whhb,
                                     bihf, bhhf, bihb, bhhb,
                                     o_wihf, o_whhf, o_wihb, o_whhb,
                                     biasf, biasb);
    k_embed<<<(Bn * Sn * En) / 256, 256, 0, stream>>>(tokens, emb, exb);
    k_proj<<<(Bn * Sn) / 16, 256, 0, stream>>>(exb, o_wihf, biasf, gatesf);
    k_proj<<<(Bn * Sn) / 16, 256, 0, stream>>>(exb, o_wihb, biasb, gatesb);
    k_lstm<<<2, 256, 0, stream>>>(gatesf, gatesb, o_whhf, o_whhb, xlen, ybf, ybfT);
    k_attn<<<Bn * (Sn / 16), 256, 0, stream>>>(ybf, ybfT, fcW, fcb, out);
}
